// sLSTMCell_54047868453103
// MI455X (gfx1250) — compile-verified, matrix operand order checked by reading, not run on hardware
//
#include <hip/hip_runtime.h>
#include <hip/hip_bf16.h>
#include <cstdint>
#include <cstddef>

#define B_SZ  4096
#define D_IN  1024
#define D_H   2048

typedef __attribute__((ext_vector_type(16))) __bf16 bf16x16;
typedef __attribute__((ext_vector_type(8)))  float  f32x8;
typedef __attribute__((ext_vector_type(4)))  unsigned int u32x4;

union BFrag {
  bf16x16 v;
  struct { u32x4 lo, hi; } u;   // lo = 8 bf16 (K..K+7), hi = 8 bf16 (K+16..K+23)
};

static __device__ __forceinline__ f32x8 wmma_bf16(bf16x16 a, bf16x16 b, f32x8 c) {
  // (neg_a, A, neg_b, B, c_mod, C, reuse_a, reuse_b)
  return __builtin_amdgcn_wmma_f32_16x16x32_bf16(false, a, false, b, (short)0, c, false, false);
}

static __device__ __forceinline__ unsigned short f2bf(float f) {
  unsigned int u = __float_as_uint(f);
  u += 0x7FFFu + ((u >> 16) & 1u);        // round to nearest even
  return (unsigned short)(u >> 16);
}

__global__ __launch_bounds__(256) void cvt_f32_to_bf16(const float* __restrict__ src,
                                                       unsigned short* __restrict__ dst,
                                                       int n) {
  int i = (blockIdx.x * blockDim.x + threadIdx.x) * 4;
  if (i >= n) return;
  const float4 f = *reinterpret_cast<const float4*>(src + i);
  uint2 o;
  o.x = (unsigned)f2bf(f.x) | ((unsigned)f2bf(f.y) << 16);
  o.y = (unsigned)f2bf(f.z) | ((unsigned)f2bf(f.w) << 16);
  *reinterpret_cast<uint2*>(dst + i) = o;
}

// Fused 4-gate GEMM (bf16 WMMA, fp32 accumulate) + sLSTM elementwise epilogue.
// Each wave computes a 64(M) x 16(N) block for all 4 gates (16 accumulator
// tiles; per K=32 step: 4 A-frags + 4 B-frags feed 16 WMMAs).
// The 8 waves of a block form a 2(M) x 4(N) grid covering 128M x 64N, so the
// A-tile is shared by 4 waves and each weight tile by 2 waves through the WGP
// cache: ~85 FLOP per L2 byte at block level.
__global__ __launch_bounds__(256) void slstm_fused_wmma(
    const unsigned short* __restrict__ xb,   // [B,  D_IN] bf16
    const unsigned short* __restrict__ hb,   // [B,  D_H ] bf16
    const unsigned short* __restrict__ Wb,   // [4*D_H, D_IN] bf16 (z,i,f,o blocks)
    const unsigned short* __restrict__ Rb,   // [4*D_H, D_H ] bf16 (z,i,f,o blocks)
    const float* __restrict__ bz, const float* __restrict__ bi,
    const float* __restrict__ bfv, const float* __restrict__ bo,
    const float* __restrict__ c_prev, const float* __restrict__ n_prev,
    float* __restrict__ hout)
{
  const int lane = threadIdx.x & 31;
  const int wave = threadIdx.x >> 5;
  const int wn   = wave & 3;      // N position within block (0..3)
  const int wm   = wave >> 2;     // M position within block (0..1)
  const int lm   = lane & 15;     // M (for A) / N (for B) within a 16-tile
  const int half = lane >> 4;     // lane-half selects K phase
  const int klo  = half * 8;

  const int n0 = blockIdx.x * 64 + wn * 16;    // wave N tile
  const int m0 = blockIdx.y * 128 + wm * 64;   // wave 64-row M block
  const int n  = n0 + lm;

  f32x8 acc[4][4];                // [gate][m_subtile]
#pragma unroll
  for (int g = 0; g < 4; ++g)
#pragma unroll
    for (int mi = 0; mi < 4; ++mi)
      acc[g][mi] = (f32x8){};

  // ---- input contraction: x @ W^T,  K = D_IN ----
  {
    const unsigned short* ap[4];
#pragma unroll
    for (int mi = 0; mi < 4; ++mi)
      ap[mi] = xb + (size_t)(m0 + mi * 16 + lm) * D_IN;
    const unsigned short* wp[4];
#pragma unroll
    for (int g = 0; g < 4; ++g)
      wp[g] = Wb + (size_t)(g * D_H + n) * D_IN;

#pragma unroll 1
    for (int k = 0; k < D_IN; k += 32) {
      BFrag a[4];
#pragma unroll
      for (int mi = 0; mi < 4; ++mi) {
        a[mi].u.lo = *reinterpret_cast<const u32x4*>(ap[mi] + k + klo);
        a[mi].u.hi = *reinterpret_cast<const u32x4*>(ap[mi] + k + klo + 16);
      }
#pragma unroll
      for (int g = 0; g < 4; ++g) {
        BFrag b;
        b.u.lo = *reinterpret_cast<const u32x4*>(wp[g] + k + klo);
        b.u.hi = *reinterpret_cast<const u32x4*>(wp[g] + k + klo + 16);
        __builtin_prefetch(wp[g] + k + 128, 0, 1);   // global_prefetch_b8
#pragma unroll
        for (int mi = 0; mi < 4; ++mi)
          acc[g][mi] = wmma_bf16(a[mi].v, b.v, acc[g][mi]);
      }
    }
  }

  // ---- recurrent contraction: h_prev @ R^T,  K = D_H ----
  {
    const unsigned short* ap[4];
#pragma unroll
    for (int mi = 0; mi < 4; ++mi)
      ap[mi] = hb + (size_t)(m0 + mi * 16 + lm) * D_H;
    const unsigned short* rp[4];
#pragma unroll
    for (int g = 0; g < 4; ++g)
      rp[g] = Rb + (size_t)(g * D_H + n) * D_H;

#pragma unroll 1
    for (int k = 0; k < D_H; k += 32) {
      BFrag a[4];
#pragma unroll
      for (int mi = 0; mi < 4; ++mi) {
        a[mi].u.lo = *reinterpret_cast<const u32x4*>(ap[mi] + k + klo);
        a[mi].u.hi = *reinterpret_cast<const u32x4*>(ap[mi] + k + klo + 16);
      }
#pragma unroll
      for (int g = 0; g < 4; ++g) {
        BFrag b;
        b.u.lo = *reinterpret_cast<const u32x4*>(rp[g] + k + klo);
        b.u.hi = *reinterpret_cast<const u32x4*>(rp[g] + k + klo + 16);
        __builtin_prefetch(rp[g] + k + 128, 0, 1);   // global_prefetch_b8
#pragma unroll
        for (int mi = 0; mi < 4; ++mi)
          acc[g][mi] = wmma_bf16(a[mi].v, b.v, acc[g][mi]);
      }
    }
  }

  // ---- epilogue: bias + gates + state update (C/D layout: VGPR r -> M = r + 8*half) ----
  const float vbz = bz[n], vbi = bi[n], vbf = bfv[n], vbo = bo[n];
#pragma unroll
  for (int mi = 0; mi < 4; ++mi) {
#pragma unroll
    for (int r = 0; r < 8; ++r) {
      const int m = m0 + mi * 16 + half * 8 + r;
      const size_t idx = (size_t)m * D_H + n;
      const float pz = acc[0][mi][r] + vbz;
      const float pi = acc[1][mi][r] + vbi;
      const float pf = acc[2][mi][r] + vbf;
      const float po = acc[3][mi][r] + vbo;
      const float zv = tanhf(pz);
      const float iv = __expf(pi);
      const float fv = __expf(pf);
      const float ov = 1.0f / (1.0f + __expf(-po));
      const float c  = fv * c_prev[idx] + iv * zv;
      const float nn = fv * n_prev[idx] + iv;
      hout[idx] = ov * (c / nn);
    }
  }
}

extern "C" void kernel_launch(void* const* d_in, const int* in_sizes, int n_in,
                              void* d_out, int out_size, void* d_ws, size_t ws_size,
                              hipStream_t stream) {
  const float* x      = (const float*)d_in[0];
  const float* h_prev = (const float*)d_in[1];
  const float* c_prev = (const float*)d_in[2];
  const float* n_prev = (const float*)d_in[3];
  const float* W[4]   = {(const float*)d_in[4], (const float*)d_in[5],
                         (const float*)d_in[6], (const float*)d_in[7]};
  const float* bz     = (const float*)d_in[8];
  const float* bi     = (const float*)d_in[9];
  const float* bf     = (const float*)d_in[10];
  const float* bo     = (const float*)d_in[11];
  const float* R[4]   = {(const float*)d_in[12], (const float*)d_in[13],
                         (const float*)d_in[14], (const float*)d_in[15]};

  // Workspace layout (bf16 as u16): xb | hb | Wb | Rb  = 8 + 16 + 16 + 32 MB = 72 MB
  unsigned short* xb = (unsigned short*)d_ws;
  unsigned short* hb = xb + (size_t)B_SZ * D_IN;
  unsigned short* Wb = hb + (size_t)B_SZ * D_H;
  unsigned short* Rb = Wb + (size_t)4 * D_H * D_IN;

  auto launch_cvt = [&](const float* s, unsigned short* d, size_t nelem) {
    int blocks = (int)((nelem / 4 + 255) / 256);
    cvt_f32_to_bf16<<<blocks, 256, 0, stream>>>(s, d, (int)nelem);
  };

  launch_cvt(x,      xb, (size_t)B_SZ * D_IN);
  launch_cvt(h_prev, hb, (size_t)B_SZ * D_H);
  for (int g = 0; g < 4; ++g)
    launch_cvt(W[g], Wb + (size_t)g * D_H * D_IN, (size_t)D_H * D_IN);
  for (int g = 0; g < 4; ++g)
    launch_cvt(R[g], Rb + (size_t)g * D_H * D_H, (size_t)D_H * D_H);

  dim3 grid(D_H / 64, B_SZ / 128);   // (32, 32)
  slstm_fused_wmma<<<grid, 256, 0, stream>>>(xb, hb, Wb, Rb,
                                             bz, bi, bf, bo,
                                             c_prev, n_prev,
                                             (float*)d_out);
}
